// MultilinearFusion_6648609374593
// MI455X (gfx1250) — matrix operand order repaired and checked by reading, not compiled, not run on hardware
//
#include <hip/hip_runtime.h>
#include <hip/hip_bf16.h>
#include <math.h>

typedef __attribute__((ext_vector_type(16))) __bf16 v16bf;
typedef __attribute__((ext_vector_type(8)))  __bf16 v8bf;
typedef __attribute__((ext_vector_type(8)))  float  v8f;
typedef __attribute__((ext_vector_type(4)))  float  f4v;

#define BATCH  64
#define D32    32
#define D1     33
#define MMHID  96
#define DFV    1185921u        // 33^4
#define OROW   (BATCH * D1)    // 2112 floats per branch output (o with appended 1)
#define NCHUNK 37061           // ceil(DFV / 32)
#define CPG    37              // K-chunks per workgroup
#define NWG    1002            // ceil(NCHUNK / CPG)
#define GSPAN  38              // max distinct T3 columns per WG slice (+slack)

struct BranchesArgs {
  const float* vh[4];
  const float* vz[4];
  const float* v3;
  const float* hw[4];
  const float* hb[4];
  const float* zw[4];
  const float* zb[4];
  const float* ow[4];
  const float* ob[4];
};

// One wave per (branch, batch-row). Computes o_br[b, 0..32] (with appended 1).
// Also zeroes the 96x64 GEMM accumulator (first 192 blocks).
__global__ __launch_bounds__(32)
void branches_kernel(BranchesArgs a, float* __restrict__ O, float* __restrict__ accb) {
  const int br = blockIdx.x & 3;
  const int b  = blockIdx.x >> 2;
  const int o  = threadIdx.x;                 // 0..31
  { int z = blockIdx.x * 32 + o; if (z < MMHID * BATCH) accb[z] = 0.f; }

  const float* vh = a.vh[br] + b * D32;
  const float* vz = a.vz[br] + b * D32;
  const float* v3 = a.v3     + b * D32;

  float h = a.hb[br][o];
  const float* hwr = a.hw[br] + o * D32;
#pragma unroll 8
  for (int i = 0; i < D32; ++i) h += vh[i] * hwr[i];
  h = fmaxf(h, 0.f);

  float z = a.zb[br][o];
  const float* zwr = a.zw[br] + o * D32 * D32;   // zw[o, i, j]
  for (int i = 0; i < D32; ++i) {
    const float* zr = zwr + i * D32;
    float s = 0.f;
#pragma unroll 8
    for (int j = 0; j < D32; ++j) s += zr[j] * v3[j];
    z += vz[i] * s;
  }
  float g = h / (1.f + __expf(-z));            // sigmoid(z) * h

  __shared__ float gs[D32];
  gs[o] = g;
  __syncthreads();

  float out = a.ob[br][o];
  const float* owr = a.ow[br] + o * D32;
#pragma unroll 8
  for (int i = 0; i < D32; ++i) out += gs[i] * owr[i];

  O[br * OROW + b * D1 + o] = fmaxf(out, 0.f);
  if (o == 0) O[br * OROW + b * D1 + D32] = 1.f;
}

// out[96,64] += W[96, DFV] x X[DFV, 64], X[f,b] = T3[b, f/33] * o4[b, f%33].
// 24 waves = 6 Mtiles x 4 Ntiles of bf16 WMMA; K split across workgroups.
// Waves 0..7 additionally materialize B tiles (8 bf16 elements/lane/chunk).
__global__ __launch_bounds__(768)
void gemm_kernel(const float* __restrict__ Wp,
                 const float* __restrict__ O,
                 float* __restrict__ accb) {
  __shared__ float ldsT3[GSPAN * 64];                 // T3 slice [g - g0][b]
  __shared__ float ldsO4x[BATCH * 66];                // o4 [b][l], duplicated (l%33)
  __shared__ __align__(32) __bf16 ldsX[2][4][32][16]; // double-buffered B tiles

  const int tid  = threadIdx.x;
  const int wave = tid >> 5;
  const int lane = tid & 31;
  const int l16  = lane & 15;
  const int half = lane >> 4;
  const int mt   = wave >> 2;                  // 0..5
  const int nt   = wave & 3;                   // 0..3

  const int c0 = blockIdx.x * CPG;
  if (c0 >= NCHUNK) return;
  const int c1 = (c0 + CPG < NCHUNK) ? c0 + CPG : NCHUNK;

  const unsigned g0 = (unsigned)(c0 * 32) / 33u;
  unsigned fend = (unsigned)(c1 * 32) - 1u;
  if (fend > DFV - 1u) fend = DFV - 1u;
  const int gcnt = (int)(fend / 33u - g0) + 1;

  // Stage o4 (duplicated to 66-wide rows) and the T3 slice into LDS.
  for (int idx = tid; idx < OROW; idx += 768) {
    int b = idx / D1, l = idx - b * D1;
    float v = O[3 * OROW + idx];
    ldsO4x[b * 66 + l]      = v;
    ldsO4x[b * 66 + 33 + l] = v;
  }
  for (int idx = tid; idx < gcnt * 64; idx += 768) {
    unsigned g = g0 + (unsigned)(idx >> 6);
    int b = idx & 63;
    unsigned i = g / 1089u;
    unsigned r = g - 1089u * i;
    unsigned j = r / 33u;
    unsigned k = r - 33u * j;
    ldsT3[idx] = O[b * D1 + (int)i] * O[OROW + b * D1 + (int)j]
               * O[2 * OROW + b * D1 + (int)k];
  }
  __syncthreads();

  const int mrow = mt * 16 + l16;
  const float* wrow  = Wp + (size_t)mrow * DFV;
  const float* lanep = wrow + 8 * half;        // per-lane A stream base
  const int bcol = nt * 16 + l16;
  const float* o4row = ldsO4x + bcol * 66;
  const float* t3col = ldsT3 + bcol;
  v8f acc = {};
  const int cm = (c1 == NCHUNK) ? (NCHUNK - 1) : c1;

  // Carried (lrun, gi) walk for former waves; f advances by 32 per chunk.
  const int fw = mt;                            // former slot (valid for wave < 8)
  unsigned fs0  = (unsigned)(c0 * 32) + 8u * (unsigned)half + 16u * (unsigned)fw;
  unsigned gcur = fs0 / 33u;
  unsigned lrun = fs0 - 33u * gcur;
  unsigned gi   = gcur - g0;

  if (wave < 8) {
    // ---- former + consumer loop ----
    for (int c = c0; c < cm; ++c) {
      const unsigned fbase = (unsigned)c * 32u;
      const float* pc = lanep + fbase;
      f4v w0, w1, w2, w3;
      __builtin_memcpy(&w0, pc,      16);
      __builtin_memcpy(&w1, pc + 4,  16);
      __builtin_memcpy(&w2, pc + 16, 16);
      __builtin_memcpy(&w3, pc + 20, 16);
      if (fbase + 1056u < DFV) __builtin_prefetch(pc + 1024, 0, 1);

      // Form 8 consecutive B elements: run spans at most 2 T3 columns.
      {
        const float* t3p = t3col + gi * 64u;
        float t3a = t3p[0], t3b = t3p[64];
        const float* o4p = o4row + lrun;
        v8bf xv;
#pragma unroll
        for (int e = 0; e < 8; ++e) {
          float x = ((lrun + (unsigned)e) < 33u ? t3a : t3b) * o4p[e];
          xv[e] = (__bf16)x;
        }
        *(v8bf*)(&ldsX[c & 1][nt][lane][8 * fw]) = xv;
        lrun += 32u; if (lrun >= 33u) { lrun -= 33u; ++gi; }
      }
      __syncthreads();

      v16bf bm = *(const v16bf*)(&ldsX[c & 1][nt][lane][0]);
      v16bf am;
#pragma unroll
      for (int e = 0; e < 4; ++e) {
        am[e] = (__bf16)w0[e]; am[4+e] = (__bf16)w1[e];
        am[8+e] = (__bf16)w2[e]; am[12+e] = (__bf16)w3[e];
      }
      acc = __builtin_amdgcn_wmma_f32_16x16x32_bf16(false, am, false, bm,
                                                    (short)0, acc, false, false);
    }
  } else {
    // ---- pure consumer loop ----
    for (int c = c0; c < cm; ++c) {
      const unsigned fbase = (unsigned)c * 32u;
      const float* pc = lanep + fbase;
      f4v w0, w1, w2, w3;
      __builtin_memcpy(&w0, pc,      16);
      __builtin_memcpy(&w1, pc + 4,  16);
      __builtin_memcpy(&w2, pc + 16, 16);
      __builtin_memcpy(&w3, pc + 20, 16);
      if (fbase + 1056u < DFV) __builtin_prefetch(pc + 1024, 0, 1);
      __syncthreads();

      v16bf bm = *(const v16bf*)(&ldsX[c & 1][nt][lane][0]);
      v16bf am;
#pragma unroll
      for (int e = 0; e < 4; ++e) {
        am[e] = (__bf16)w0[e]; am[4+e] = (__bf16)w1[e];
        am[8+e] = (__bf16)w2[e]; am[12+e] = (__bf16)w3[e];
      }
      acc = __builtin_amdgcn_wmma_f32_16x16x32_bf16(false, am, false, bm,
                                                    (short)0, acc, false, false);
    }
  }

  // Peeled final chunk (only the last WG): clamped A loads, zero-padded B.
  if (c1 == NCHUNK) {
    const int c = NCHUNK - 1;
    const unsigned fbase = (unsigned)c * 32u;
    float a0[8], a1[8];
#pragma unroll
    for (int v = 0; v < 8; ++v) {
      unsigned Kb = 2u * (unsigned)v + (v >= 4 ? 8u : 0u) + 8u * (unsigned)half;
      unsigned f0 = fbase + Kb;
      unsigned fa = (f0      < DFV) ? f0      : DFV - 1u;
      unsigned fb = (f0 + 1u < DFV) ? f0 + 1u : DFV - 1u;
      a0[v] = wrow[fa];
      a1[v] = wrow[fb];
    }
    if (wave < 8) {
      const float* t3p = t3col + gi * 64u;
      float t3a = t3p[0], t3b = t3p[64];
      const float* o4p = o4row + lrun;
      unsigned fs = fbase + 8u * (unsigned)half + 16u * (unsigned)fw;
      v8bf xv;
#pragma unroll
      for (int e = 0; e < 8; ++e) {
        unsigned f = fs + (unsigned)e;
        float x = 0.f;
        if (f < DFV) x = ((lrun + (unsigned)e) < 33u ? t3a : t3b) * o4p[e];
        xv[e] = (__bf16)x;
      }
      *(v8bf*)(&ldsX[c & 1][nt][lane][8 * fw]) = xv;
    }
    __syncthreads();
    v16bf bm = *(const v16bf*)(&ldsX[c & 1][nt][lane][0]);
    v16bf am;
#pragma unroll
    for (int v = 0; v < 8; ++v) { am[2 * v] = (__bf16)a0[v]; am[2 * v + 1] = (__bf16)a1[v]; }
    acc = __builtin_amdgcn_wmma_f32_16x16x32_bf16(false, am, false, bm,
                                                  (short)0, acc, false, false);
  }

  const int ncol = nt * 16 + l16;
#pragma unroll
  for (int r = 0; r < 8; ++r) {
    int mr = mt * 16 + r + 8 * half;   // D layout: VGPR r, lanes16-31 -> M+8
    atomicAdd(&accb[mr * 64 + ncol], acc[r]);
  }
}

// relu(acc + enc1_b), concat skip [o1..o4], enc2, relu.
__global__ __launch_bounds__(256)
void head_kernel(const float* __restrict__ accb, const float* __restrict__ O,
                 const float* __restrict__ b1, const float* __restrict__ w2,
                 const float* __restrict__ b2, float* __restrict__ out) {
  int idx = blockIdx.x * 256 + threadIdx.x;
  if (idx >= BATCH * MMHID) return;
  int b = idx / MMHID;
  int m = idx - b * MMHID;
  const float* wr = w2 + m * (MMHID + 4 * D1);
  float s = b2[m];
#pragma unroll 4
  for (int c = 0; c < MMHID; ++c)
    s += wr[c] * fmaxf(accb[c * 64 + b] + b1[c], 0.f);
#pragma unroll 4
  for (int t = 0; t < 4 * D1; ++t) {
    int br = t / D1, e = t - br * D1;
    s += wr[MMHID + t] * O[br * OROW + b * D1 + e];
  }
  out[idx] = fmaxf(s, 0.f);
}

extern "C" void kernel_launch(void* const* d_in, const int* in_sizes, int n_in,
                              void* d_out, int out_size, void* d_ws, size_t ws_size,
                              hipStream_t stream) {
  const float* vec1 = (const float*)d_in[0];
  const float* vec2 = (const float*)d_in[1];
  const float* vec3 = (const float*)d_in[2];
  const float* vec4 = (const float*)d_in[3];
  const float* P[24];
  for (int i = 0; i < 24; ++i) P[i] = (const float*)d_in[4 + i];
  const float* enc1_w = (const float*)d_in[28];
  const float* enc1_b = (const float*)d_in[29];
  const float* enc2_w = (const float*)d_in[30];
  const float* enc2_b = (const float*)d_in[31];

  float* ws   = (float*)d_ws;
  float* O    = ws;               // 4 * 2112 floats
  float* accb = ws + 4 * OROW;    // 96 * 64 floats

  BranchesArgs a;
  const float* vhs[4] = {vec1, vec2, vec3, vec4};
  const float* vzs[4] = {vec1, vec2, vec1, vec4};  // branch3 gate uses (vec1, vec3)
  for (int br = 0; br < 4; ++br) {
    a.vh[br] = vhs[br];      a.vz[br] = vzs[br];
    a.hw[br] = P[6 * br + 0]; a.hb[br] = P[6 * br + 1];
    a.zw[br] = P[6 * br + 2]; a.zb[br] = P[6 * br + 3];
    a.ow[br] = P[6 * br + 4]; a.ob[br] = P[6 * br + 5];
  }
  a.v3 = vec3;

  branches_kernel<<<4 * BATCH, 32, 0, stream>>>(a, O, accb);
  gemm_kernel<<<NWG, 768, 0, stream>>>(enc1_w, O, accb);
  head_kernel<<<(BATCH * MMHID + 255) / 256, 256, 0, stream>>>(
      accb, O, enc1_b, enc2_w, enc2_b, (float*)d_out);
}
